// _HessianLinear_39779987096002
// MI455X (gfx1250) — matrix operand order repaired and checked by reading, not compiled, not run on hardware
//
#include <hip/hip_runtime.h>

// ---------------------------------------------------------------------------
// Quantized linear:  out[m,n] = sum_k x[m,k] * (centroids[labels[n,k]] /
//                    max(scale[k],1e-8))  + bias[n]
// M = B*S = 8192, N = OUT_F = 4096, K = IN_F = 4096  -> 275 GFLOP GEMM
// (compute-bound at bf16 WMMA rate; min HBM traffic ~350 MB ≈ 15 us).
// Strategy: dequantize W to bf16 (32 MB) and pack x to bf16 (64 MB) into
// workspace; both stay resident in the 192 MB L2, so the tiled WMMA GEMM's
// operand re-reads never touch HBM.  Matrix core: v_wmma_f32_16x16x32_bf16.
// Global->LDS staging uses CDNA5 GLOBAL_LOAD_ASYNC_TO_LDS_B128 (ASYNCcnt),
// keeping the VGPR file and VMEM-return path free for WMMA operands.
// ---------------------------------------------------------------------------

typedef __bf16 bf16_t;
typedef __attribute__((ext_vector_type(16))) __bf16 v16bf;
typedef __attribute__((ext_vector_type(8)))  __bf16 v8bf;
typedef __attribute__((ext_vector_type(4)))  __bf16 v4bf;
typedef __attribute__((ext_vector_type(8)))  float  v8f;
typedef __attribute__((ext_vector_type(4)))  float  v4f;
typedef __attribute__((ext_vector_type(4)))  int    v4i;

#define OUT_F 4096
#define IN_F  4096
#define MTOT  8192          // B*S = 4*2048

#define BM 128              // block tile M
#define BN 128              // block tile N
#define BK 64               // staged k-tile = 2 WMMA K-steps (K=32 each)

// --- CDNA5 async global->LDS copy (16B per lane), tracked by ASYNCcnt ------
__device__ __forceinline__ void async_copy_b128(unsigned lds_byte_off,
                                                const void* gaddr) {
  asm volatile("global_load_async_to_lds_b128 %0, %1, off"
               :: "v"(lds_byte_off), "v"(gaddr)
               : "memory");
}

__device__ __forceinline__ void wait_asynccnt0() {
#if __has_builtin(__builtin_amdgcn_s_wait_asynccnt)
  __builtin_amdgcn_s_wait_asynccnt(0);
#else
  asm volatile("s_wait_asynccnt 0x0" ::: "memory");
#endif
}

// __shared__ pointer -> LDS byte offset (low 32 bits of the flat LDS address)
__device__ __forceinline__ unsigned lds_off(const void* p) {
  return (unsigned)(unsigned long long)p;
}

// ---------------------------------------------------------------------------
// Kernel 1: W_bf16[o,k] = bf16( centroids[labels[o,k]] / max(scale[k],1e-8) )
// ---------------------------------------------------------------------------
__global__ __launch_bounds__(256)
void dequant_w_kernel(const float* __restrict__ centroids,
                      const int*   __restrict__ labels,
                      const float* __restrict__ scale,
                      bf16_t*      __restrict__ Wq) {
  const long base = ((long)blockIdx.x * 256 + threadIdx.x) * 4;
  const v4i lab = *(const v4i*)(labels + base);
  const v4f sc  = *(const v4f*)(scale + (int)(base & (long)(IN_F - 1)));
  v4bf o;
#pragma unroll
  for (int j = 0; j < 4; ++j) {
    float s = sc[j] > 1e-8f ? sc[j] : 1e-8f;
    o[j] = (bf16_t)(centroids[lab[j]] / s);
  }
  *(v4bf*)(Wq + base) = o;
}

// ---------------------------------------------------------------------------
// Kernel 2: x f32 -> bf16 pack (halves activation footprint; 64 MB fits L2).
// ---------------------------------------------------------------------------
__global__ __launch_bounds__(256)
void cvt_x_kernel(const float* __restrict__ x, bf16_t* __restrict__ Xq) {
  const long base = ((long)blockIdx.x * 256 + threadIdx.x) * 8;
  const v4f a = *(const v4f*)(x + base);
  const v4f b = *(const v4f*)(x + base + 4);
  v8bf o;
#pragma unroll
  for (int j = 0; j < 4; ++j) { o[j] = (bf16_t)a[j]; o[j + 4] = (bf16_t)b[j]; }
  *(v8bf*)(Xq + base) = o;
}

// ---------------------------------------------------------------------------
// Kernel 3: bf16 WMMA GEMM.  out = Xq(8192x4096) * Wq(4096x4096)^T + bias.
// 256 threads = 8 waves; block tile 128x128; wave tile 64(M) x 32(N)
// = 4x2 accumulators of 16x16 f32.  LDS double-buffered (BK=64 staged tile
// = 2 WMMA K-steps -> 16 WMMAs per barrier per wave), async-copy fill.
// ---------------------------------------------------------------------------
__global__ __launch_bounds__(256)
void gemm_bf16_kernel(const bf16_t* __restrict__ Xq,   // (MTOT, IN_F)
                      const bf16_t* __restrict__ Wq,   // (OUT_F, IN_F)
                      const float*  __restrict__ bias, // (OUT_F)
                      float*        __restrict__ out)  // (MTOT, OUT_F)
{
  __shared__ __align__(16) bf16_t lA[2][BM * BK];   // 16 KB per buffer
  __shared__ __align__(16) bf16_t lB[2][BN * BK];   // 16 KB per buffer

  const int tid  = threadIdx.x;
  const int lane = tid & 31;
  const int wave = tid >> 5;
  const int wm   = wave >> 2;                 // 0..1  -> M offset wm*64
  const int wn   = wave & 3;                  // 0..3  -> N offset wn*32

  const int mTile = blockIdx.y * BM;
  const int nTile = blockIdx.x * BN;

  // --- staging assignment: each thread owns 32 contiguous bf16 (64 B) of
  //     one row of the X tile and of the W tile -> 4 async b128 copies each.
  const int lrow  = tid >> 1;                 // 0..127
  const int lhalf = (tid & 1) * 32;           // 0 or 32 (elements in K)
  const bf16_t* gA = Xq + (long)(mTile + lrow) * IN_F + lhalf;
  const bf16_t* gB = Wq + (long)(nTile + lrow) * IN_F + lhalf;
  const unsigned wrA0 = lds_off(&lA[0][lrow * BK + lhalf]);
  const unsigned wrB0 = lds_off(&lB[0][lrow * BK + lhalf]);
  const unsigned bufStrideA = (unsigned)(BM * BK * sizeof(bf16_t));
  const unsigned bufStrideB = (unsigned)(BN * BK * sizeof(bf16_t));

  // --- per-lane LDS read offsets, contiguous K order k = 16*lhi + s (same for A and B) -------------
  const int l15 = lane & 15;
  const int lhi = lane >> 4;
  // A 16-bit 16x32: lane m<16 holds K 0..15 ; lane m+16 holds
  // K 16..31 -> chunks at +lhi*16 and +lhi*16+8 within a k-step.
  const int aOffBase = (wm * 64 + l15) * BK + lhi * 16;
  // B 16-bit 32x16: lanes 0-15 hold K=0..15 of column n, lanes 16-31 hold
  // K=16..31 -> 16 contiguous K per lane (= contiguous row chunk of Wq).
  const int bOffBase = (wn * 32 + l15) * BK + lhi * 16;

  const v8f vzero = {0.f, 0.f, 0.f, 0.f, 0.f, 0.f, 0.f, 0.f};
  v8f acc[4][2];
#pragma unroll
  for (int mi = 0; mi < 4; ++mi)
#pragma unroll
    for (int ni = 0; ni < 2; ++ni) acc[mi][ni] = vzero;

  // ---- preload k-tile 0 into buffer 0 (async, direct to LDS) -------------
#pragma unroll
  for (int c = 0; c < 4; ++c) {
    async_copy_b128(wrA0 + c * 16, gA + c * 8);
    async_copy_b128(wrB0 + c * 16, gB + c * 8);
  }
  wait_asynccnt0();
  __syncthreads();

  int buf = 0;
  for (int kk = 0; kk < IN_F; kk += BK, buf ^= 1) {
    const bool has_next = (kk + BK) < IN_F;

    // ---- issue async fill of the other buffer (overlaps WMMA below) -----
    if (has_next) {
      const unsigned wa = wrA0 + (buf ^ 1) * bufStrideA;
      const unsigned wb = wrB0 + (buf ^ 1) * bufStrideB;
#pragma unroll
      for (int c = 0; c < 4; ++c) {
        async_copy_b128(wa + c * 16, gA + kk + BK + c * 8);
        async_copy_b128(wb + c * 16, gB + kk + BK + c * 8);
      }
    }

    // ---- compute current staged tile: 2 WMMA K-steps x (4x2) tiles ------
    const bf16_t* Ab = &lA[buf][0];
    const bf16_t* Bb = &lB[buf][0];
#pragma unroll
    for (int h = 0; h < 2; ++h) {             // k-step within staged tile
      const int kh = h * 32;
      v16bf bfrag[2];
#pragma unroll
      for (int ni = 0; ni < 2; ++ni) {
        const int off = bOffBase + ni * 16 * BK + kh;
        v8bf c0 = *(const v8bf*)(Bb + off);
        v8bf c1 = *(const v8bf*)(Bb + off + 8);
        bfrag[ni] = __builtin_shufflevector(c0, c1, 0, 1, 2, 3, 4, 5, 6, 7, 8,
                                            9, 10, 11, 12, 13, 14, 15);
      }
#pragma unroll
      for (int mi = 0; mi < 4; ++mi) {
        const int off = aOffBase + mi * 16 * BK + kh;
        v8bf c0 = *(const v8bf*)(Ab + off);
        v8bf c1 = *(const v8bf*)(Ab + off + 8);
        v16bf afrag = __builtin_shufflevector(c0, c1, 0, 1, 2, 3, 4, 5, 6, 7,
                                              8, 9, 10, 11, 12, 13, 14, 15);
        acc[mi][0] = __builtin_amdgcn_wmma_f32_16x16x32_bf16(
            false, afrag, false, bfrag[0], (short)0, acc[mi][0], false, false);
        acc[mi][1] = __builtin_amdgcn_wmma_f32_16x16x32_bf16(
            false, afrag, false, bfrag[1], (short)0, acc[mi][1], false, false);
      }
    }

    // ---- make the freshly filled buffer visible to all waves ------------
    if (has_next) wait_asynccnt0();
    __syncthreads();
  }

  // ---- epilogue: C/D layout (VGPR v: lanes0-15 M=v, lanes16-31 M=v+8) ----
  const int mBase = mTile + wm * 64 + lhi * 8;
  const int nBase = nTile + wn * 32;
#pragma unroll
  for (int ni = 0; ni < 2; ++ni) {
    const int n0 = nBase + ni * 16 + l15;
    const float bv = bias[n0];
#pragma unroll
    for (int mi = 0; mi < 4; ++mi) {
      const int m0 = mBase + mi * 16;
#pragma unroll
      for (int v = 0; v < 8; ++v) {
        out[(long)(m0 + v) * OUT_F + n0] = acc[mi][ni][v] + bv;
      }
    }
  }
}

// ---------------------------------------------------------------------------
// Launcher.  Inputs (setup_inputs order): x f32[8192*4096],
// centroids f32[256], labels i32[4096*4096], scale f32[4096], bias f32[4096].
// Workspace layout: [0, 32MB) Wq bf16 ; [32MB, 96MB) Xq bf16.
// ---------------------------------------------------------------------------
extern "C" void kernel_launch(void* const* d_in, const int* in_sizes, int n_in,
                              void* d_out, int out_size, void* d_ws,
                              size_t ws_size, hipStream_t stream) {
  const float* x         = (const float*)d_in[0];
  const float* centroids = (const float*)d_in[1];
  const int*   labels    = (const int*)d_in[2];
  const float* scale     = (const float*)d_in[3];
  const float* bias      = (const float*)d_in[4];
  float* out = (float*)d_out;

  bf16_t* Wq = (bf16_t*)d_ws;                                     // 32 MB
  bf16_t* Xq = (bf16_t*)((char*)d_ws + (size_t)OUT_F * IN_F * 2); // 64 MB

  // 1) dequantize weights -> bf16 (16.7M elems, 4/thread)
  dequant_w_kernel<<<(OUT_F * (long)IN_F) / (256 * 4), 256, 0, stream>>>(
      centroids, labels, scale, Wq);

  // 2) pack activations -> bf16 (33.5M elems, 8/thread)
  cvt_x_kernel<<<((long)MTOT * IN_F) / (256 * 8), 256, 0, stream>>>(x, Xq);

  // 3) WMMA GEMM, grid = (N/128, M/128) = (32, 64)
  dim3 grid(OUT_F / BN, MTOT / BM);
  gemm_bf16_kernel<<<grid, 256, 0, stream>>>(Xq, Wq, bias, out);
}
